// Head_17712445129204
// MI455X (gfx1250) — matrix operand order repaired and checked by reading, not compiled, not run on hardware
//
#include <hip/hip_runtime.h>
#include <hip/hip_bf16.h>

typedef _Float16 v16h __attribute__((ext_vector_type(16)));
typedef _Float16 v8h  __attribute__((ext_vector_type(8)));
typedef float    v8f  __attribute__((ext_vector_type(8)));

union U16H { v16h v; v8h h[2]; };

#define C_DIM 400
#define HDIM  64

#define WMMA_F16(a, b, c) \
  __builtin_amdgcn_wmma_f32_16x16x32_f16(false, (a), false, (b), (short)0, (c), false, false)

// ---------------------------------------------------------------------------
// Stage 1: q = x@Wq * (scale*log2e) -> f16 [B*T,64]
//          k = x@Wk                 -> f16 [B*T,64]   (head-dim contiguous)
//          vT = (x@Wv)^T            -> f16 [B,64,T]   (key-dim contiguous)
// 4 token rows per block: each weight load from L2 feeds 4 FMAs.
// ---------------------------------------------------------------------------
__global__ __launch_bounds__(64) void qkv_proj_kernel(
    const float* __restrict__ x, const float* __restrict__ Wq,
    const float* __restrict__ Wk, const float* __restrict__ Wv,
    _Float16* __restrict__ qh, _Float16* __restrict__ kh,
    _Float16* __restrict__ vT, int T)
{
  __shared__ float xs[4][C_DIM];
  const int row0 = blockIdx.x * 4;
  const int h    = threadIdx.x;          // 0..63
  const float* xrows = x + (size_t)row0 * C_DIM;
  for (int i = h; i < 4 * C_DIM; i += 64) ((float*)xs)[i] = xrows[i];
  __syncthreads();

  float aq[4] = {}, ak[4] = {}, av[4] = {};
  for (int c = 0; c < C_DIM; ++c) {
    const float wq = Wq[c * HDIM + h];   // coalesced across threads
    const float wk = Wk[c * HDIM + h];
    const float wv = Wv[c * HDIM + h];
    #pragma unroll
    for (int r = 0; r < 4; ++r) {
      const float xv = xs[r][c];
      aq[r] = fmaf(xv, wq, aq[r]);
      ak[r] = fmaf(xv, wk, ak[r]);
      av[r] = fmaf(xv, wv, av[r]);
    }
  }
  const float qs = 0.125f * 1.44269504088896340736f;  // 1/sqrt(64) * log2(e)
  #pragma unroll
  for (int r = 0; r < 4; ++r) {
    const int row = row0 + r;
    qh[(size_t)row * HDIM + h] = (_Float16)(aq[r] * qs);
    kh[(size_t)row * HDIM + h] = (_Float16)ak[r];
    const int b = row / T, t = row - b * T;
    vT[((size_t)b * HDIM + h) * T + t] = (_Float16)av[r];
  }
}

// ---------------------------------------------------------------------------
// Stage 2: FlashAttention. Block = 128 threads (4 waves), 16 query rows/wave.
// Software-pipelined K loads; row sums via WMMA against all-ones B matrix.
// ---------------------------------------------------------------------------
__global__ __launch_bounds__(128) void flash_attn_kernel(
    const _Float16* __restrict__ qh, const _Float16* __restrict__ kh,
    const _Float16* __restrict__ vT, float* __restrict__ out, int T)
{
  __shared__ _Float16 pbuf[4][16][40];   // per-wave 16x32 P tile, 16B-aligned rows

  const int lane = threadIdx.x & 31;
  const int wave = threadIdx.x >> 5;
  const int tb   = blockIdx.x & 31;      // T/64 = 32 row-tiles
  const int b    = blockIdx.x >> 5;
  const int row0 = tb * 64 + wave * 16;  // wave's first query row
  const int half = lane >> 4;
  const int ln   = lane & 15;

  const _Float16* __restrict__ qrow  = qh + ((size_t)b * T + row0 + ln) * HDIM;
  const _Float16* __restrict__ kbase = kh + (size_t)b * T * HDIM;
  const _Float16* __restrict__ vbase = vT + (size_t)b * HDIM * T;

  // Loop-invariant Q A-operands (head-dim blocks hb=0,32).
  U16H qa[2];
  {
    const int ka = half * 16;
    #pragma unroll
    for (int hb = 0; hb < 2; ++hb) {
      qa[hb].h[0] = *(const v8h*)(qrow + hb * 32 + ka);
      qa[hb].h[1] = *(const v8h*)(qrow + hb * 32 + 8 + ka);
    }
  }

  // All-ones B operand: osum += P . ones gives per-row sums in C layout.
  v16h ones;
  #pragma unroll
  for (int i = 0; i < 16; ++i) ones[i] = (_Float16)1.0f;

  v8f o[4] = {};       // O accumulators (4 h-tiles), C layout
  v8f osum = {};       // running softmax denominators, C layout
  float rmax[8];
  #pragma unroll
  for (int r = 0; r < 8; ++r) rmax[r] = -3.0e38f;

  auto load_kb = [&](int kt, v16h kb[2][2]) {
    #pragma unroll
    for (int nt = 0; nt < 2; ++nt) {
      const int key = kt + nt * 16 + ln;
      kb[nt][0] = *(const v16h*)(kbase + (size_t)key * HDIM +  0 + half * 16);
      kb[nt][1] = *(const v16h*)(kbase + (size_t)key * HDIM + 32 + half * 16);
    }
  };

  const int kt_end = tb * 64 + 63;       // block-uniform trip count
  v16h kb[2][2];
  load_kb(0, kb);                        // kt0 = 0 is active for every wave

  for (int kt0 = 0; kt0 <= kt_end; kt0 += 32) {
    const bool active     = (kt0 <= row0 + 15);        // wave-uniform
    const bool nextActive = (kt0 + 32 <= row0 + 15);   // wave-uniform

    v16h vb[4];
    v8f  s[2];
    if (active) {
      // Issue V loads early; consumed only after softmax + LDS transpose.
      #pragma unroll
      for (int ht = 0; ht < 4; ++ht)
        vb[ht] = *(const v16h*)(vbase + (size_t)(ht * 16 + ln) * T + kt0 + half * 16);

      // S[16x32] = Q(16x64) . K^T(64x32)
      #pragma unroll
      for (int nt = 0; nt < 2; ++nt) {
        v8f acc = {};
        acc = WMMA_F16(qa[0].v, kb[nt][0], acc);
        acc = WMMA_F16(qa[1].v, kb[nt][1], acc);
        s[nt] = acc;
      }
    }

    // Pipeline: issue next tile's K loads; latency hides behind softmax.
    v16h kbn[2][2];
    if (nextActive) {
      load_kb(kt0 + 32, kbn);
      if (kt0 + 64 <= row0 + 15)
        __builtin_prefetch(kbase + (size_t)(kt0 + 64 + lane) * HDIM, 0, 3);
    }

    float alpha[8];
    if (active) {
      // Causal mask only on diagonal tiles (wave-uniform branch).
      if (kt0 + 31 > row0) {
        #pragma unroll
        for (int nt = 0; nt < 2; ++nt) {
          const int key = kt0 + nt * 16 + ln;
          #pragma unroll
          for (int r = 0; r < 8; ++r)
            if (key > row0 + r + 8 * half) s[nt][r] = -3.0e38f;
        }
      }
      // Online softmax in log2 domain (scale*log2e folded into Q).
      #pragma unroll
      for (int r = 0; r < 8; ++r) {
        float tmax = fmaxf(s[0][r], s[1][r]);
        #pragma unroll
        for (int m = 1; m < 16; m <<= 1)       // stays within 16-lane half
          tmax = fmaxf(tmax, __shfl_xor(tmax, m, 32));
        const float nmax = fmaxf(rmax[r], tmax);
        alpha[r] = exp2f(rmax[r] - nmax);
        rmax[r]  = nmax;
        s[0][r] = exp2f(s[0][r] - nmax);
        s[1][r] = exp2f(s[1][r] - nmax);
      }
      #pragma unroll
      for (int r = 0; r < 8; ++r) {
        osum[r] *= alpha[r];
        #pragma unroll
        for (int ht = 0; ht < 4; ++ht) o[ht][r] *= alpha[r];
      }
    }

    // P: C-layout -> A-layout via LDS (barriers are block-uniform).
    __syncthreads();
    if (active) {
      #pragma unroll
      for (int nt = 0; nt < 2; ++nt)
        #pragma unroll
        for (int r = 0; r < 8; ++r)
          pbuf[wave][r + 8 * half][nt * 16 + ln] = (_Float16)s[nt][r];
    }
    __syncthreads();

    if (active) {
      U16H pa;
      const int ka = half * 16;
      pa.h[0] = *(const v8h*)(&pbuf[wave][ln][ka]);
      pa.h[1] = *(const v8h*)(&pbuf[wave][ln][8 + ka]);

      // O += P(16x32) . V(32x64); denominators += P . ones
      #pragma unroll
      for (int ht = 0; ht < 4; ++ht)
        o[ht] = WMMA_F16(pa.v, vb[ht], o[ht]);
      osum = WMMA_F16(pa.v, ones, osum);
    }

    if (nextActive) {
      #pragma unroll
      for (int nt = 0; nt < 2; ++nt) {
        kb[nt][0] = kbn[nt][0];
        kb[nt][1] = kbn[nt][1];
      }
    }
  }

  // Normalize and write fp32 output.
  float* __restrict__ obase = out + ((size_t)b * T + row0) * HDIM;
  #pragma unroll
  for (int r = 0; r < 8; ++r) {
    const float inv = 1.0f / osum[r];
    #pragma unroll
    for (int ht = 0; ht < 4; ++ht)
      obase[(size_t)(r + 8 * half) * HDIM + ht * 16 + ln] = o[ht][r] * inv;
  }
}

// ---------------------------------------------------------------------------
extern "C" void kernel_launch(void* const* d_in, const int* in_sizes, int n_in,
                              void* d_out, int out_size, void* d_ws, size_t ws_size,
                              hipStream_t stream)
{
  const float* x  = (const float*)d_in[0];
  const float* Wq = (const float*)d_in[1];
  const float* Wk = (const float*)d_in[2];
  const float* Wv = (const float*)d_in[3];

  const int T  = 2048;
  const int BT = in_sizes[0] / C_DIM;   // B*T
  const int B  = BT / T;

  size_t arr = ((size_t)BT * HDIM * sizeof(_Float16) + 255) & ~(size_t)255;
  _Float16* qh = (_Float16*)d_ws;
  _Float16* kh = (_Float16*)((char*)d_ws + arr);
  _Float16* vT = (_Float16*)((char*)d_ws + 2 * arr);

  qkv_proj_kernel<<<BT / 4, 64, 0, stream>>>(x, Wq, Wk, Wv, qh, kh, vT, T);
  flash_attn_kernel<<<B * (T / 64), 128, 0, stream>>>(qh, kh, vT, (float*)d_out, T);
}
